// KANFeedForward_58626303591100
// MI455X (gfx1250) — matrix operand order repaired and checked by reading, not compiled, not run on hardware
//
#include <hip/hip_runtime.h>

// ---------------------------------------------------------------------------
// KAN feed-forward for gfx1250 (MI455X), two-phase fused WMMA GEMM per layer:
//   phase 1 (spline): C  = Bases(x) @ spline_w^T   with K = IN*8
//       aug A[m][i*8+c] = B_c(x[m][i])   (8 cubic B-spline bases per scalar,
//       computed once per scalar via a shared Cox-de-Boor triangle)
//       aug W[o][i*8+c] = spline_w[o][i][c]  == spline_w flattened (contiguous!)
//   phase 2 (base):   C += SiLU(x) @ base_w^T      with K = IN
// Both phases accumulate into the same v_wmma_f32_16x16x32_bf16 accumulators.
// x strips are staged to LDS with GLOBAL_LOAD_ASYNC_TO_LDS (ASYNCcnt) when the
// toolchain exposes the builtin.
// ---------------------------------------------------------------------------

typedef __attribute__((ext_vector_type(16))) __bf16 v16bf;
typedef __attribute__((ext_vector_type(8)))  __bf16 v8bf;
typedef __attribute__((ext_vector_type(4)))  __bf16 v4bf;
typedef __attribute__((ext_vector_type(8)))  float  v8f;
typedef __attribute__((ext_vector_type(4)))  float  v4f;

#if __has_builtin(__builtin_amdgcn_global_load_async_to_lds_b32)
#define KAN_ASYNC_LDS 1
typedef __attribute__((address_space(1))) int kan_gint;
typedef __attribute__((address_space(3))) int kan_lint;
#endif

__device__ __forceinline__ void kan_wait_async() {
#if defined(KAN_ASYNC_LDS)
#if __has_builtin(__builtin_amdgcn_s_wait_asynccnt)
    __builtin_amdgcn_s_wait_asynccnt(0);
#else
    asm volatile("s_wait_asynccnt 0x0" ::: "memory");
#endif
#endif
}

__device__ __forceinline__ float silu_f(float x) {
    return x * (1.f / (1.f + __expf(-x)));
}

// All 8 cubic B-spline bases on the uniform knot grid t_m = 0.4*m - 2.2
// (efficient-kan grid, GRID_SIZE=5, SPLINE_ORDER=3): one shared triangle.
__device__ __forceinline__ void bspline_all8(float x, float out[8]) {
    float b[11];
#pragma unroll
    for (int i = 0; i < 11; ++i) {
        const float ti = 0.4f * (float)i - 2.2f;
        b[i] = (x >= ti && x < ti + 0.4f) ? 1.f : 0.f;
    }
#pragma unroll
    for (int j = 1; j <= 3; ++j) {
        const float inv = 1.f / (0.4f * (float)j);
#pragma unroll
        for (int i = 0; i < 11 - j; ++i) {
            const float ti = 0.4f * (float)i - 2.2f;
            const float left  = (x - ti) * inv;
            const float right = ((ti + 0.4f * (float)(j + 1)) - x) * inv;
            b[i] = left * b[i] + right * b[i + 1];
        }
    }
#pragma unroll
    for (int i = 0; i < 8; ++i) out[i] = b[i];
}

__device__ __forceinline__ v16bf cat8(v8bf lo, v8bf hi) {
    return __builtin_shufflevector(lo, hi, 0, 1, 2, 3, 4, 5, 6, 7,
                                           8, 9, 10, 11, 12, 13, 14, 15);
}

// BM=BN=128, BK=64 (2 WMMA K-steps per staged chunk), 8 wave32 per block.
template <int IN>
__global__ __launch_bounds__(256) void kan_gemm(
    const float* __restrict__ X,        // (M, IN)
    const float* __restrict__ baseW,    // (Nout, IN)
    const float* __restrict__ splineW,  // (Nout, IN*8) contiguous
    float* __restrict__ C,              // (M, Nout)
    int M, int Nout) {
    constexpr int BM = 128, BN = 128, BK = 64;
    constexpr int LD = BK + 8;                 // 144B row stride (16B multiple)

    __shared__ __align__(16) __bf16 As[BM * LD];
    __shared__ __align__(16) __bf16 Bs[BN * LD];
    __shared__ float Xs[BM * (BK / 8)];        // 128 x 8 f32 strip (spline)

    const int tid   = threadIdx.x;
    const int m0    = blockIdx.y * BM;
    const int n0    = blockIdx.x * BN;
    const int wave  = tid >> 5;
    const int lane  = tid & 31;
    const int wm    = (wave & 1) * 64;         // 2 waves along M, 4 tiles each
    const int wn    = (wave >> 1) * 32;        // 4 waves along N, 2 tiles each
    const int lhalf = lane >> 4;
    const int lmod  = lane & 15;

    v8f acc[4][2] = {};

    auto mma_chunk = [&]() {
#pragma unroll
        for (int kk = 0; kk < BK; kk += 32) {
            v16bf af[4];
            const int akb = kk + lhalf * 8;    // A: lanes 16-31 hold K+8/K+24
#pragma unroll
            for (int mt = 0; mt < 4; ++mt) {
                const __bf16* p = &As[(wm + mt * 16 + lmod) * LD + akb];
                af[mt] = cat8(*(const v8bf*)p, *(const v8bf*)(p + 16));
            }
            v16bf bfr[2];
            const int bkb = kk + lhalf * 8;    // B: lanes 16-31 hold K+8/K+24
#pragma unroll
            for (int nt = 0; nt < 2; ++nt) {
                const __bf16* p = &Bs[(wn + nt * 16 + lmod) * LD + bkb];
                bfr[nt] = cat8(*(const v8bf*)p, *(const v8bf*)(p + 16));
            }
#pragma unroll
            for (int mt = 0; mt < 4; ++mt)
#pragma unroll
                for (int nt = 0; nt < 2; ++nt)
                    acc[mt][nt] = __builtin_amdgcn_wmma_f32_16x16x32_bf16(
                        false, af[mt], false, bfr[nt],
                        (short)0, acc[mt][nt], false, false);
        }
    };

    // ---------------- phase 1: spline path, K = IN*8 --------------------
    const int K1 = IN * 8;
    for (int k0 = 0; k0 < K1; k0 += BK) {
        const int i0 = k0 >> 3;                // first of 8 x-columns
        __syncthreads();
        // x strip -> LDS (async-to-LDS when available)
#pragma unroll
        for (int t = 0; t < 4; ++t) {
            const int e = tid + t * 256;       // [0, 1024)
            const int m = e >> 3, g = e & 7;
            const float* gp = &X[(size_t)(m0 + m) * IN + i0 + g];
#if defined(KAN_ASYNC_LDS)
            __builtin_amdgcn_global_load_async_to_lds_b32(
                (kan_gint*)gp, (kan_lint*)&Xs[e], 0, 0);
#else
            Xs[e] = *gp;
#endif
        }
        // B tile: spline weights are contiguous f32; float4 loads -> v8bf
#pragma unroll
        for (int t = 0; t < 4; ++t) {
            const int e = tid + t * 256;       // [0, 1024) groups of 8
            const int n = e >> 3, g = e & 7;
            const float* wp = &splineW[(size_t)(n0 + n) * K1 + k0 + g * 8];
            __builtin_prefetch(wp + BK, 0, 1); // next K-chunk hint
            const v4f w0 = *(const v4f*)wp;
            const v4f w1 = *(const v4f*)(wp + 4);
            v8bf o;
            o[0] = (__bf16)w0[0]; o[1] = (__bf16)w0[1];
            o[2] = (__bf16)w0[2]; o[3] = (__bf16)w0[3];
            o[4] = (__bf16)w1[0]; o[5] = (__bf16)w1[1];
            o[6] = (__bf16)w1[2]; o[7] = (__bf16)w1[3];
            *(v8bf*)&Bs[n * LD + g * 8] = o;
        }
        kan_wait_async();
        __syncthreads();
        // A tile: 8 bases per scalar, one aligned 16B LDS store each
#pragma unroll
        for (int t = 0; t < 4; ++t) {
            const int e = tid + t * 256;
            const int m = e >> 3, g = e & 7;
            float bb[8];
            bspline_all8(Xs[e], bb);
            v8bf o;
#pragma unroll
            for (int c = 0; c < 8; ++c) o[c] = (__bf16)bb[c];
            *(v8bf*)&As[m * LD + g * 8] = o;
        }
        __syncthreads();
        mma_chunk();
    }

    // ---------------- phase 2: base path, K = IN ------------------------
    for (int k0 = 0; k0 < IN; k0 += BK) {
        __syncthreads();
#pragma unroll
        for (int t = 0; t < 8; ++t) {
            const int e = tid + t * 256;       // [0, 2048) float4 groups
            const int m = e >> 4, q = e & 15;
            const v4f xv = *(const v4f*)&X[(size_t)(m0 + m) * IN + k0 + q * 4];
            v4bf o;
            o[0] = (__bf16)silu_f(xv[0]); o[1] = (__bf16)silu_f(xv[1]);
            o[2] = (__bf16)silu_f(xv[2]); o[3] = (__bf16)silu_f(xv[3]);
            *(v4bf*)&As[m * LD + q * 4] = o;
        }
#pragma unroll
        for (int t = 0; t < 8; ++t) {
            const int e = tid + t * 256;
            const int n = e >> 4, q = e & 15;
            const v4f wv = *(const v4f*)&baseW[(size_t)(n0 + n) * IN + k0 + q * 4];
            v4bf o;
            o[0] = (__bf16)wv[0]; o[1] = (__bf16)wv[1];
            o[2] = (__bf16)wv[2]; o[3] = (__bf16)wv[3];
            *(v4bf*)&Bs[n * LD + q * 4] = o;
        }
        __syncthreads();
        mma_chunk();
    }

    // ---- epilogue: C/D layout = VGPR r -> M = r + 8*lhalf, N = lmod ----
#pragma unroll
    for (int mt = 0; mt < 4; ++mt)
#pragma unroll
        for (int nt = 0; nt < 2; ++nt)
#pragma unroll
            for (int r = 0; r < 8; ++r) {
                const int row = m0 + wm + mt * 16 + r + lhalf * 8;
                const int col = n0 + wn + nt * 16 + lmod;
                C[(size_t)row * Nout + col] = acc[mt][nt][r];
            }
}

extern "C" void kernel_launch(void* const* d_in, const int* in_sizes, int n_in,
                              void* d_out, int out_size, void* d_ws, size_t ws_size,
                              hipStream_t stream) {
    constexpr int D_IN = 1024, D_HID = 4096, D_OUT = 1024;
    const float* x   = (const float*)d_in[0];
    const float* bw1 = (const float*)d_in[1];
    const float* sw1 = (const float*)d_in[2];  // (D_HID, D_IN*8) contiguous
    const float* bw2 = (const float*)d_in[3];
    const float* sw2 = (const float*)d_in[4];  // (D_OUT, D_HID*8) contiguous
    // d_in[5], d_in[6]: uniform grids -> compile-time constants in the math.

    const int M = in_sizes[0] / D_IN;          // B*S = 2048
    float* h   = (float*)d_ws;                 // (M, D_HID) f32, 33.5 MB
    float* out = (float*)d_out;

    dim3 blk(256);
    kan_gemm<D_IN><<<dim3(D_HID / 128, M / 128), blk, 0, stream>>>(
        x, bw1, sw1, h, M, D_HID);
    kan_gemm<D_HID><<<dim3(D_OUT / 128, M / 128), blk, 0, stream>>>(
        h, bw2, sw2, out, M, D_OUT);
}
